// SinkhornDistance_20401094656240
// MI455X (gfx1250) — compile-verified, matrix operand order checked by reading, not run on hardware
//
#include <hip/hip_runtime.h>

// Problem constants (from reference): N=8, P1=P2=2048, eps=0.1, 50 iters.
#define NB   8
#define P1   2048
#define P2   2048
#define NIT  50

// exp/log in base-2 domain: z = (x)/eps * log2(e)
#define K2      14.426950408889634f   // log2(e)/eps
#define EPSLN2  0.06931471805599453f  // eps*ln(2)
#define EPSV    0.1f
#define LOGEPS  1e-8f

// Raw hardware transcendentals (v_exp_f32 / v_log_f32). All our exp args are
// <= 0 (post max-subtraction), where raw v_exp_f32 is exact and flushes large
// negatives to 0 -- exactly the LSE semantics. Avoids the ~6-VALU safe-range
// expansion of exp2f.
#if __has_builtin(__builtin_amdgcn_exp2f)
#define EXP2(x) __builtin_amdgcn_exp2f(x)
#else
#define EXP2(x) exp2f(x)
#endif
#if __has_builtin(__builtin_amdgcn_logf)
#define LOG2(x) __builtin_amdgcn_logf(x)
#else
#define LOG2(x) __log2f(x)
#endif

typedef __attribute__((ext_vector_type(16))) _Float16 v16h;
typedef __attribute__((ext_vector_type(8)))  float    v8f;

// ---------------------------------------------------------------------------
// init: u=v=0, elmu = eps*log(mu+1e-8), elnu = eps*log(nu+1e-8), cost=0
// ---------------------------------------------------------------------------
__global__ void __launch_bounds__(256) sink_init(const float* __restrict__ mu,
                                                 const float* __restrict__ nu,
                                                 float* __restrict__ u,
                                                 float* __restrict__ v,
                                                 float* __restrict__ elmu,
                                                 float* __restrict__ elnu,
                                                 float* __restrict__ cost) {
  int t = blockIdx.x * blockDim.x + threadIdx.x;
  if (t < NB) cost[t] = 0.0f;
  if (t < NB * P1) {
    u[t] = 0.0f;
    elmu[t] = EPSV * __logf(mu[t] + LOGEPS);
  }
  if (t < NB * P2) {
    v[t] = 0.0f;
    elnu[t] = EPSV * __logf(nu[t] + LOGEPS);
  }
}

// ---------------------------------------------------------------------------
// u' = elmu - eps*ln2*( m2 + log2( sum_j 2^{z-m2} ) ),  z = (v_j - C_ij)*K2
// One block = 16 rows x 2048 cols.  Sum-of-exp via V_WMMA with B = ones.
// ---------------------------------------------------------------------------
__global__ void __launch_bounds__(256) u_update(const float* __restrict__ C,
                                                const float* __restrict__ v,
                                                const float* __restrict__ elmu,
                                                float* __restrict__ u) {
  const int n  = blockIdx.y;
  const int i0 = blockIdx.x * 16;
  const int tid = threadIdx.x;

  __shared__ float s_red[16][17];
  __shared__ float s_m2[16];
  __shared__ float s_part[8][16];

  const float* Cb = C + ((size_t)n * P1 + i0) * P2;
  const float* vb = v + (size_t)n * P2;

  // ---- phase 1: per-row max of z (base-2 domain), float4 coalesced ----
  const int ry = tid >> 4;         // row 0..15
  const int cc = tid & 15;         // 16 threads per row, float4 each
  const float* rowC = Cb + (size_t)ry * P2;
  float lm = -3.0e38f;
  for (int j = cc * 4; j < P2; j += 64) {
    float4 cv = *(const float4*)(rowC + j);
    float4 vv = *(const float4*)(vb + j);
    lm = fmaxf(lm, (vv.x - cv.x) * K2);
    lm = fmaxf(lm, (vv.y - cv.y) * K2);
    lm = fmaxf(lm, (vv.z - cv.z) * K2);
    lm = fmaxf(lm, (vv.w - cv.w) * K2);
  }
  s_red[ry][cc] = lm;
  __syncthreads();
  if (tid < 16) {
    float m = s_red[tid][0];
#pragma unroll
    for (int k = 1; k < 16; ++k) m = fmaxf(m, s_red[tid][k]);
    s_m2[tid] = m;
  }
  __syncthreads();

  // ---- phase 2: row sums of 2^(z-m2) via WMMA (B = ones) ----
  const int wave = tid >> 5;
  const int lane = tid & 31;
  const int r    = lane & 15;            // A-matrix row this lane feeds
  const int kb   = (lane >> 4) * 8;      // K half-select per 16x32 f16 A layout
  const float  m2r = s_m2[r];
  const float* rC  = Cb + (size_t)r * P2;

  v16h ones;
#pragma unroll
  for (int h = 0; h < 16; ++h) ones[h] = (_Float16)1.0f;

  v8f acc = {};
#pragma unroll
  for (int c = 0; c < 8; ++c) {
    const int K0 = wave * 256 + c * 32 + kb;   // multiple of 8 -> 32B aligned
    // two 8-float runs per lane: [K0, K0+8) and [K0+16, K0+24)
    const float4* rc4 = (const float4*)(rC + K0);
    const float4* vv4 = (const float4*)(vb + K0);
    float4 ca = rc4[0], cb = rc4[1], cd = rc4[4], ce = rc4[5];
    float4 va = vv4[0], vvb = vv4[1], vd = vv4[4], ve = vv4[5];

    float e0  = EXP2((va.x  - ca.x) * K2 - m2r);
    float e1  = EXP2((va.y  - ca.y) * K2 - m2r);
    float e2  = EXP2((va.z  - ca.z) * K2 - m2r);
    float e3  = EXP2((va.w  - ca.w) * K2 - m2r);
    float e4  = EXP2((vvb.x - cb.x) * K2 - m2r);
    float e5  = EXP2((vvb.y - cb.y) * K2 - m2r);
    float e6  = EXP2((vvb.z - cb.z) * K2 - m2r);
    float e7  = EXP2((vvb.w - cb.w) * K2 - m2r);
    float e8  = EXP2((vd.x  - cd.x) * K2 - m2r);
    float e9  = EXP2((vd.y  - cd.y) * K2 - m2r);
    float e10 = EXP2((vd.z  - cd.z) * K2 - m2r);
    float e11 = EXP2((vd.w  - cd.w) * K2 - m2r);
    float e12 = EXP2((ve.x  - ce.x) * K2 - m2r);
    float e13 = EXP2((ve.y  - ce.y) * K2 - m2r);
    float e14 = EXP2((ve.z  - ce.z) * K2 - m2r);
    float e15 = EXP2((ve.w  - ce.w) * K2 - m2r);

    v16h a;
    a[0]  = (_Float16)e0;  a[1]  = (_Float16)e1;
    a[2]  = (_Float16)e2;  a[3]  = (_Float16)e3;
    a[4]  = (_Float16)e4;  a[5]  = (_Float16)e5;
    a[6]  = (_Float16)e6;  a[7]  = (_Float16)e7;
    a[8]  = (_Float16)e8;  a[9]  = (_Float16)e9;
    a[10] = (_Float16)e10; a[11] = (_Float16)e11;
    a[12] = (_Float16)e12; a[13] = (_Float16)e13;
    a[14] = (_Float16)e14; a[15] = (_Float16)e15;

    // D = A x ones + acc  -> every column of D holds the running row sums
    acc = __builtin_amdgcn_wmma_f32_16x16x32_f16(
        /*neg_a=*/false, a, /*neg_b=*/false, ones,
        /*c_mod=*/(short)0, acc, /*reuse_a=*/false, /*reuse_b=*/false);
  }
  // C/D layout: lane 0 (N=0) VGPR g -> row g ; lane 16 (N=0) VGPR g -> row g+8
  if (lane == 0) {
#pragma unroll
    for (int g = 0; g < 8; ++g) s_part[wave][g] = acc[g];
  }
  if (lane == 16) {
#pragma unroll
    for (int g = 0; g < 8; ++g) s_part[wave][8 + g] = acc[g];
  }
  __syncthreads();

  if (tid < 16) {
    float s = 0.0f;
#pragma unroll
    for (int w = 0; w < 8; ++w) s += s_part[w][tid];
    const size_t gi = (size_t)n * P1 + i0 + tid;
    u[gi] = elmu[gi] - EPSLN2 * (s_m2[tid] + LOG2(s));
  }
}

// ---------------------------------------------------------------------------
// v' = elnu - eps*ln2*LSE2_i( (u_i - C_ij)*K2 )
// Block = 64 columns (16 x float4) x 16 row-splits (128 rows each).
// Coalesced 512B/wave row reads; 4 independent online-LSE states per thread.
// ---------------------------------------------------------------------------
__global__ void __launch_bounds__(256) v_update(const float* __restrict__ C,
                                                const float* __restrict__ u,
                                                const float* __restrict__ elnu,
                                                float* __restrict__ v) {
  const int n  = blockIdx.y;
  const int jb = blockIdx.x * 64;
  const int tid = threadIdx.x;
  const int c = tid & 15;      // float4 column group within tile (4 cols)
  const int s = tid >> 4;      // row split 0..15 (128 rows each)

  __shared__ float s_m[16][16][4];
  __shared__ float s_s[16][16][4];

  const float* Cb = C + (size_t)n * P1 * P2 + jb + c * 4;
  const float* ub = u + (size_t)n * P1;

  float m[4], sum[4];
#pragma unroll
  for (int k = 0; k < 4; ++k) { m[k] = -3.0e38f; sum[k] = 0.0f; }

  const int i0 = s * 128, i1 = i0 + 128;
  for (int i = i0; i < i1; ++i) {
    float4 cv = *(const float4*)(Cb + (size_t)i * P2);
    const float ui = ub[i];
    float z[4] = { (ui - cv.x) * K2, (ui - cv.y) * K2,
                   (ui - cv.z) * K2, (ui - cv.w) * K2 };
#pragma unroll
    for (int k = 0; k < 4; ++k) {
      float nm = fmaxf(m[k], z[k]);
      sum[k] = sum[k] * EXP2(m[k] - nm) + EXP2(z[k] - nm);
      m[k] = nm;
    }
  }
#pragma unroll
  for (int k = 0; k < 4; ++k) { s_m[s][c][k] = m[k]; s_s[s][c][k] = sum[k]; }
  __syncthreads();

  if (tid < 16) {
#pragma unroll
    for (int k = 0; k < 4; ++k) {
      float M = s_m[0][tid][k];
#pragma unroll
      for (int p = 1; p < 16; ++p) M = fmaxf(M, s_m[p][tid][k]);
      float S = 0.0f;
#pragma unroll
      for (int p = 0; p < 16; ++p) S += s_s[p][tid][k] * EXP2(s_m[p][tid][k] - M);
      const size_t gj = (size_t)n * P2 + jb + tid * 4 + k;
      v[gj] = elnu[gj] - EPSLN2 * (M + LOG2(S));
    }
  }
}

// ---------------------------------------------------------------------------
// final: pi = 2^((u+v-C)*K2), copy C, cost[n] += sum(pi*C)  (float4 pass)
// ---------------------------------------------------------------------------
__global__ void __launch_bounds__(256) pi_cost(const float* __restrict__ C,
                                               const float* __restrict__ u,
                                               const float* __restrict__ v,
                                               float* __restrict__ outCost,
                                               float* __restrict__ outPi,
                                               float* __restrict__ outC) {
  const int n = blockIdx.y;
  const size_t base = ((size_t)blockIdx.x * 256 + threadIdx.x) * 4; // within batch
  const size_t gb = (size_t)n * P1 * P2 + base;
  const int i = (int)(base >> 11);     // /2048
  const int j = (int)(base & 2047);

  float4 c4 = *(const float4*)(C + gb);
  const float ui = u[(size_t)n * P1 + i];
  float4 v4 = *(const float4*)(v + (size_t)n * P2 + j);

  float4 p4;
  p4.x = EXP2((ui + v4.x - c4.x) * K2);
  p4.y = EXP2((ui + v4.y - c4.y) * K2);
  p4.z = EXP2((ui + v4.z - c4.z) * K2);
  p4.w = EXP2((ui + v4.w - c4.w) * K2);

  *(float4*)(outPi + gb) = p4;
  *(float4*)(outC  + gb) = c4;

  float acc = p4.x * c4.x + p4.y * c4.y + p4.z * c4.z + p4.w * c4.w;
  __shared__ float red[256];
  red[threadIdx.x] = acc;
  __syncthreads();
  for (int st = 128; st > 0; st >>= 1) {
    if (threadIdx.x < st) red[threadIdx.x] += red[threadIdx.x + st];
    __syncthreads();
  }
  if (threadIdx.x == 0) atomicAdd(&outCost[n], red[0]);
}

// ---------------------------------------------------------------------------
extern "C" void kernel_launch(void* const* d_in, const int* in_sizes, int n_in,
                              void* d_out, int out_size, void* d_ws, size_t ws_size,
                              hipStream_t stream) {
  (void)in_sizes; (void)n_in; (void)out_size; (void)ws_size;
  const float* C  = (const float*)d_in[0];
  const float* mu = (const float*)d_in[1];
  const float* nu = (const float*)d_in[2];

  float* ws   = (float*)d_ws;
  float* u    = ws;
  float* v    = u + (size_t)NB * P1;
  float* elmu = v + (size_t)NB * P2;
  float* elnu = elmu + (size_t)NB * P1;

  float* cost  = (float*)d_out;
  float* outPi = cost + NB;
  float* outC  = outPi + (size_t)NB * P1 * P2;

  sink_init<<<dim3((NB * P1 + 255) / 256), 256, 0, stream>>>(mu, nu, u, v, elmu, elnu, cost);

  for (int it = 0; it < NIT; ++it) {
    u_update<<<dim3(P1 / 16, NB), 256, 0, stream>>>(C, v, elmu, u);
    v_update<<<dim3(P2 / 64, NB), 256, 0, stream>>>(C, u, elnu, v);
  }

  pi_cost<<<dim3((P1 * P2) / (256 * 4), NB), 256, 0, stream>>>(C, u, v, cost, outPi, outC);
}
